// PolaLinearAttention_6502580486630
// MI455X (gfx1250) — matrix operand order repaired and checked by reading, not compiled, not run on hardware
//
#include <hip/hip_runtime.h>
#include <hip/hip_bf16.h>
#include <math.h>

// Problem constants (match reference)
#define B_  4
#define N_  4096
#define C_  512
#define H_  8
#define HD_ 64

typedef __attribute__((ext_vector_type(16))) __bf16 v16bf;
typedef __attribute__((ext_vector_type(8)))  float  v8f;

__device__ __forceinline__ unsigned short f32_to_bf16_rne(float f) {
    union { float f; unsigned u; } x; x.f = f;
    unsigned u = x.u;
    u += 0x7FFFu + ((u >> 16) & 1u);   // round-to-nearest-even
    return (unsigned short)(u >> 16);
}

__device__ __forceinline__ unsigned lds_addr32(const void* p) {
    // shared-aperture flat address: low 32 bits are the LDS byte offset
    return (unsigned)(size_t)p;
}

__device__ __forceinline__ void async_cp16(unsigned lds_off, const void* gaddr) {
    // 16B global -> LDS DMA, tracked by ASYNCcnt
    asm volatile("global_load_async_to_lds_b128 %0, %1, off"
                 :: "v"(lds_off), "v"(gaddr) : "memory");
}

__device__ __forceinline__ void wait_async0() {
    asm volatile("s_wait_asynccnt 0x0" ::: "memory");
}

// ---------------------------------------------------------------------------
// fp32 -> bf16 pack (for A operands)
// ---------------------------------------------------------------------------
__global__ __launch_bounds__(256)
void pack_bf16_kernel(const float* __restrict__ in, unsigned short* __restrict__ out)
{
    int idx = blockIdx.x * 256 + threadIdx.x;
    out[idx] = f32_to_bf16_rne(in[idx]);
}

// fp32 W[K,Nc] -> bf16 WT[Nc,K] (transposed so B-tile staging is a plain copy)
__global__ __launch_bounds__(256)
void transpose_pack_kernel(const float* __restrict__ W, unsigned short* __restrict__ WT,
                           int K, int Nc)
{
    int idx = blockIdx.x * 256 + threadIdx.x;   // over Nc*K
    int n = idx / K, k = idx % K;
    WT[idx] = f32_to_bf16_rne(W[(size_t)k * Nc + n]);
}

// P = (xo + vc) * g packed to bf16  (g = qg[..., C:], row stride 2C)
__global__ __launch_bounds__(256)
void combine_pack_kernel(const float* __restrict__ xo, const float* __restrict__ vc,
                         const float* __restrict__ qg, unsigned short* __restrict__ Pbf)
{
    int idx = blockIdx.x * 256 + threadIdx.x;   // over B*N*C
    int row = idx / C_, c = idx % C_;
    float g = qg[(size_t)row * (2 * C_) + C_ + c];
    Pbf[idx] = f32_to_bf16_rne((xo[idx] + vc[idx]) * g);
}

// ---------------------------------------------------------------------------
// bf16 WMMA GEMM with async-LDS double-buffered staging.
//   Out[M,Nc] = Abf[M,K] @ WT[Nc,K]^T  (+bias)
// Block = 256 threads = 8 waves, tile 128x128, K-step 32.
// LDS rows padded to 40 shorts (80B): 16B-aligned chunks, conflict-free
// (20-bank row stride). Fragments load as 2 x ds_load_b128 each.
// ---------------------------------------------------------------------------
__global__ __launch_bounds__(256)
void gemm_wmma_bf16_async(const unsigned short* __restrict__ Abf,
                          const unsigned short* __restrict__ WT,
                          const float* __restrict__ bias,
                          float* __restrict__ Out,
                          int M, int K, int Ncols)
{
    __shared__ __align__(16) unsigned short As[2][128 * 40];
    __shared__ __align__(16) unsigned short Bs[2][128 * 40];

    const int t        = threadIdx.x;
    const int col_base = blockIdx.x * 128;
    const int row_base = blockIdx.y * 128;
    const int wave  = t >> 5;
    const int lane  = t & 31;
    const int l16   = lane & 15;
    const int khalf = lane >> 4;

    const unsigned asA0 = lds_addr32(&As[0][0]);
    const unsigned asB0 = lds_addr32(&Bs[0][0]);
    const unsigned bufStride = 128 * 40 * 2;   // bytes per buffer

    v8f acc[8] = {};
    const int nk = K / 32;

    // issue the 16B async copies for one K-tile (A: 128x32 bf16, B: 128x32 bf16)
    auto prefetch = [&](int kt, int buf) {
        #pragma unroll
        for (int i = 0; i < 2; ++i) {
            int c  = t + i * 256;       // 0..511 chunks
            int r  = c >> 2;            // tile row 0..127
            int cp = c & 3;             // 16B chunk within 64B row
            size_t ga = ((size_t)(row_base + r) * K + kt) * 2 + cp * 16;
            size_t gb = ((size_t)(col_base + r) * K + kt) * 2 + cp * 16;
            unsigned lofs = (unsigned)(r * 80 + cp * 16) + (unsigned)buf * bufStride;
            async_cp16(asA0 + lofs, (const char*)Abf + ga);
            async_cp16(asB0 + lofs, (const char*)WT + gb);
        }
    };

    prefetch(0, 0);
    wait_async0();
    __syncthreads();

    for (int i = 0; i < nk; ++i) {
        int buf = i & 1;
        if (i + 1 < nk) prefetch((i + 1) * 32, buf ^ 1);

        // A fragment (16-bit A 16x32 layout): 2 x 16B LDS loads
        v16bf af;
        {
            const unsigned short* arow = &As[buf][(wave * 16 + l16) * 40];
            uint4* afp = reinterpret_cast<uint4*>(&af);
            afp[0] = *reinterpret_cast<const uint4*>(arow + 8 * khalf);
            afp[1] = *reinterpret_cast<const uint4*>(arow + 16 + 8 * khalf);
        }
        // all 8 B fragments first (2 x 16B each), then 8 back-to-back WMMAs
        v16bf bfr[8];
        #pragma unroll
        for (int j = 0; j < 8; ++j) {
            const unsigned short* brow = &Bs[buf][(j * 16 + l16) * 40];
            uint4* bfp = reinterpret_cast<uint4*>(&bfr[j]);
            bfp[0] = *reinterpret_cast<const uint4*>(brow + 8 * khalf);
            bfp[1] = *reinterpret_cast<const uint4*>(brow + 16 + 8 * khalf);
        }
        #pragma unroll
        for (int j = 0; j < 8; ++j)
            acc[j] = __builtin_amdgcn_wmma_f32_16x16x32_bf16(
                false, af, false, bfr[j], (short)0, acc[j], false, false);

        if (i + 1 < nk) wait_async0();
        __syncthreads();
    }

    // epilogue: C layout VGPR v -> row v + 8*khalf, col = l16
    #pragma unroll
    for (int j = 0; j < 8; ++j) {
        int col = col_base + j * 16 + l16;
        float bb = bias ? bias[col] : 0.0f;
        #pragma unroll
        for (int v = 0; v < 8; ++v) {
            int row = row_base + wave * 16 + v + 8 * khalf;
            Out[(size_t)row * Ncols + col] = acc[j][v] + bb;
        }
    }
}

// ---------------------------------------------------------------------------
// Elementwise: split qg/kv, softplus scale, sigmoid power, pow_relu features.
// Outputs head-major (B,H,N,HD).
// ---------------------------------------------------------------------------
__global__ __launch_bounds__(256)
void transform_kernel(const float* __restrict__ qg, const float* __restrict__ kv,
                      const float* __restrict__ pos_enc,
                      const float* __restrict__ scale_p,
                      const float* __restrict__ power_p,
                      float* __restrict__ qp, float* __restrict__ qn,
                      float* __restrict__ kp, float* __restrict__ kn,
                      float* __restrict__ vh)
{
    int idx = blockIdx.x * 256 + threadIdx.x;   // over B*N*C
    int c  = idx % C_;
    int rn = idx / C_;
    int n  = rn % N_;
    int b  = rn / N_;
    int h  = c / HD_;
    int hd = c % HD_;

    size_t qgo = ((size_t)b * N_ + n) * (2 * C_) + c;
    float q = qg[qgo];
    float k = kv[qgo] + pos_enc[(size_t)n * C_ + c];
    float v = kv[qgo + C_];

    float s     = scale_p[c];
    float scale = fmaxf(s, 0.0f) + log1pf(expf(-fabsf(s)));      // softplus
    float pw    = 1.0f + 4.0f / (1.0f + expf(-power_p[h * HD_ + hd]));

    float qs = q / scale;
    float ks = k / scale;

    size_t o = (((size_t)b * H_ + h) * N_ + n) * HD_ + hd;
    qp[o] = qs > 0.0f ? powf(qs, pw)  : 0.0f;
    qn[o] = qs < 0.0f ? powf(-qs, pw) : 0.0f;
    kp[o] = ks > 0.0f ? powf(ks, pw)  : 0.0f;
    kn[o] = ks < 0.0f ? powf(-ks, pw) : 0.0f;
    vh[o] = v;
}

// ---------------------------------------------------------------------------
// kmean[bh][d] = mean_n kc[n][d],  kc = [kp | kn] (d in [0,128))
// ---------------------------------------------------------------------------
__global__ __launch_bounds__(128)
void kmean_kernel(const float* __restrict__ kp, const float* __restrict__ kn,
                  float* __restrict__ kmean)
{
    int bh = blockIdx.x;          // 0..31
    int d  = threadIdx.x;         // 0..127
    const float* src = (d < HD_ ? kp : kn) + (size_t)bh * N_ * HD_ + (d & (HD_ - 1));
    float s = 0.0f;
    for (int n = 0; n < N_; ++n) s += src[(size_t)n * HD_];
    kmean[bh * 128 + d] = s * (1.0f / (float)N_);
}

// ---------------------------------------------------------------------------
// kvm[bh][d][e] = (1/N) * sum_n kc[n][d] * vh[n][e]   (d<128, e<64)
// ---------------------------------------------------------------------------
__global__ __launch_bounds__(256)
void kvm_kernel(const float* __restrict__ kp, const float* __restrict__ kn,
                const float* __restrict__ vh, float* __restrict__ kvm)
{
    __shared__ float kcs[32][128];
    __shared__ float vs[32][64];
    int bh = blockIdx.x;
    int t  = threadIdx.x;
    int d  = t & 127;
    int e0 = (t >> 7) * 32;
    float acc[32] = {};
    size_t base = (size_t)bh * N_ * HD_;

    for (int n0 = 0; n0 < N_; n0 += 32) {
        #pragma unroll
        for (int i = 0; i < 16; ++i) {
            int li = i * 256 + t;
            int nn = li >> 7, dd = li & 127;
            const float* kk = (dd < HD_ ? kp : kn);
            kcs[nn][dd] = kk[base + (size_t)(n0 + nn) * HD_ + (dd & (HD_ - 1))];
        }
        #pragma unroll
        for (int i = 0; i < 8; ++i) {
            int li = i * 256 + t;
            int nn = li >> 6, ee = li & 63;
            vs[nn][ee] = vh[base + (size_t)(n0 + nn) * HD_ + ee];
        }
        __syncthreads();
        for (int nn = 0; nn < 32; ++nn) {
            float kc = kcs[nn][d];
            #pragma unroll
            for (int e = 0; e < 32; ++e) acc[e] += kc * vs[nn][e0 + e];
        }
        __syncthreads();
    }
    float rn2 = 1.0f / (float)N_;
    for (int e = 0; e < 32; ++e)
        kvm[((size_t)bh * 128 + d) * 64 + e0 + e] = acc[e] * rn2;
}

// ---------------------------------------------------------------------------
// x_sim/x_opp with fused z normalization. Block = (b,h) x 64 rows of n.
// ---------------------------------------------------------------------------
__global__ __launch_bounds__(256)
void xapply_kernel(const float* __restrict__ qp, const float* __restrict__ qn,
                   const float* __restrict__ kvm, const float* __restrict__ kmean,
                   float* __restrict__ xo)
{
    __shared__ float kvmS[128 * 64];
    __shared__ float kmS[128];
    __shared__ float qs[4][128];
    __shared__ float zs[4][2];

    const int nblocks = N_ / 64;
    int bh = blockIdx.x / nblocks;
    int nb = blockIdx.x % nblocks;
    int t  = threadIdx.x;
    int b  = bh / H_, h = bh % H_;

    #pragma unroll
    for (int i = 0; i < 32; ++i)
        kvmS[i * 256 + t] = kvm[(size_t)bh * 8192 + i * 256 + t];
    if (t < 128) kmS[t] = kmean[bh * 128 + t];
    __syncthreads();

    int nl = t >> 6;
    int cc = t & 63;
    size_t qbase0 = ((size_t)bh * N_ + nb * 64) * HD_;

    for (int it = 0; it < 16; ++it) {
        int nofs = it * 4 + nl;
        size_t qb = qbase0 + (size_t)nofs * HD_;
        qs[nl][cc]      = qp[qb + cc];
        qs[nl][64 + cc] = qn[qb + cc];
        __syncthreads();
        if (cc == 0) {
            float s1 = 0.0f, s2 = 0.0f;
            for (int d = 0; d < 64; ++d) {
                float a  = qs[nl][d];
                float bq = qs[nl][64 + d];
                s1 += a * kmS[d] + bq * kmS[64 + d];
                s2 += bq * kmS[d] + a * kmS[64 + d];
            }
            zs[nl][0] = 1.0f / (s1 + 1e-6f);
            zs[nl][1] = 1.0f / (s2 + 1e-6f);
        }
        __syncthreads();
        bool sim = cc < 32;
        int qo1 = sim ? 0 : 64;
        int qo2 = sim ? 64 : 0;
        float v = 0.0f;
        #pragma unroll 8
        for (int d = 0; d < 64; ++d)
            v += qs[nl][qo1 + d] * kvmS[d * 64 + cc]
               + qs[nl][qo2 + d] * kvmS[(64 + d) * 64 + cc];
        v *= zs[nl][sim ? 0 : 1];
        int n = nb * 64 + nofs;
        xo[((size_t)b * N_ + n) * C_ + h * HD_ + cc] = v;
        __syncthreads();
    }
}

// ---------------------------------------------------------------------------
// Depthwise 5x5 conv, 8x8 tile + halo in LDS, channel-innermost.
// ---------------------------------------------------------------------------
__global__ __launch_bounds__(256)
void dwconv_kernel(const float* __restrict__ vh, const float* __restrict__ w,
                   const float* __restrict__ bias, float* __restrict__ vc)
{
    __shared__ float vt[12][12][64];
    int img  = blockIdx.x >> 6;
    int tile = blockIdx.x & 63;
    int ty0  = (tile >> 3) * 8;
    int tx0  = (tile & 7) * 8;
    int t    = threadIdx.x;
    size_t base = (size_t)img * N_ * HD_;

    #pragma unroll
    for (int i = 0; i < 36; ++i) {
        int li = i * 256 + t;
        int ch = li & 63;
        int p  = li >> 6;
        int yy = p / 12, xx = p % 12;
        int gy = ty0 + yy - 2, gx = tx0 + xx - 2;
        float v = 0.0f;
        if (gy >= 0 && gy < 64 && gx >= 0 && gx < 64)
            v = vh[base + (size_t)(gy * 64 + gx) * HD_ + ch];
        vt[yy][xx][ch] = v;
    }
    __syncthreads();

    int b = img / H_, h = img % H_;
    #pragma unroll
    for (int i = 0; i < 16; ++i) {
        int li = i * 256 + t;
        int ch = li & 63;
        int p  = li >> 6;
        int oy = p >> 3, ox = p & 7;
        float acc = bias[ch];
        #pragma unroll
        for (int dy = 0; dy < 5; ++dy)
            #pragma unroll
            for (int dx = 0; dx < 5; ++dx)
                acc += w[ch * 25 + dy * 5 + dx] * vt[oy + dy][ox + dx][ch];
        int n = (ty0 + oy) * 64 + (tx0 + ox);
        vc[((size_t)b * N_ + n) * C_ + h * HD_ + ch] = acc;
    }
}

// ---------------------------------------------------------------------------
extern "C" void kernel_launch(void* const* d_in, const int* in_sizes, int n_in,
                              void* d_out, int out_size, void* d_ws, size_t ws_size,
                              hipStream_t stream)
{
    (void)in_sizes; (void)n_in; (void)out_size; (void)ws_size;

    const float* x       = (const float*)d_in[0];
    const float* Wqg     = (const float*)d_in[1];
    const float* Wkv     = (const float*)d_in[2];
    const float* Wproj   = (const float*)d_in[3];
    const float* bproj   = (const float*)d_in[4];
    const float* dwc_w   = (const float*)d_in[5];
    const float* dwc_b   = (const float*)d_in[6];
    const float* power_p = (const float*)d_in[7];
    const float* scale_p = (const float*)d_in[8];
    const float* pos_enc = (const float*)d_in[9];
    float* out = (float*)d_out;

    // workspace carving (in floats; every region stays 16B-aligned:
    // all counts are multiples of 4 floats)
    float* ws = (float*)d_ws;
    size_t o = 0;
    float* qg    = ws + o; o += (size_t)B_ * N_ * 2 * C_;
    float* kvb   = ws + o; o += (size_t)B_ * N_ * 2 * C_;
    float* qp    = ws + o; o += (size_t)B_ * H_ * N_ * HD_;
    float* qn    = ws + o; o += (size_t)B_ * H_ * N_ * HD_;
    float* kp    = ws + o; o += (size_t)B_ * H_ * N_ * HD_;
    float* kn    = ws + o; o += (size_t)B_ * H_ * N_ * HD_;
    float* vh    = ws + o; o += (size_t)B_ * H_ * N_ * HD_;
    float* kmean = ws + o; o += (size_t)B_ * H_ * 128;
    float* kvm   = ws + o; o += (size_t)B_ * H_ * 128 * 64;
    float* xo    = ws + o; o += (size_t)B_ * N_ * C_;
    float* vc    = ws + o; o += (size_t)B_ * N_ * C_;
    // bf16 regions (2 shorts per float slot)
    unsigned short* xbf    = (unsigned short*)(ws + o); o += (size_t)B_ * N_ * C_ / 2;
    unsigned short* WqgT   = (unsigned short*)(ws + o); o += (size_t)2 * C_ * C_ / 2;
    unsigned short* WkvT   = (unsigned short*)(ws + o); o += (size_t)2 * C_ * C_ / 2;
    unsigned short* WprojT = (unsigned short*)(ws + o); o += (size_t)C_ * C_ / 2;
    unsigned short* Pbf    = (unsigned short*)(ws + o); o += (size_t)B_ * N_ * C_ / 2;

    dim3 blk(256);
    const int M = B_ * N_;   // 16384

    // 0. pack operands to bf16 (A row-major, W transposed to [Nc,K])
    pack_bf16_kernel<<<(B_ * N_ * C_) / 256, blk, 0, stream>>>(x, xbf);
    transpose_pack_kernel<<<(2 * C_ * C_) / 256, blk, 0, stream>>>(Wqg, WqgT, C_, 2 * C_);
    transpose_pack_kernel<<<(2 * C_ * C_) / 256, blk, 0, stream>>>(Wkv, WkvT, C_, 2 * C_);
    transpose_pack_kernel<<<(C_ * C_) / 256, blk, 0, stream>>>(Wproj, WprojT, C_, C_);

    // 1-2. qg = x@Wqg, kv = x@Wkv  (async-LDS bf16 WMMA)
    gemm_wmma_bf16_async<<<dim3((2 * C_) / 128, M / 128), blk, 0, stream>>>(
        xbf, WqgT, nullptr, qg, M, C_, 2 * C_);
    gemm_wmma_bf16_async<<<dim3((2 * C_) / 128, M / 128), blk, 0, stream>>>(
        xbf, WkvT, nullptr, kvb, M, C_, 2 * C_);

    // 3. elementwise features
    transform_kernel<<<(B_ * N_ * C_) / 256, blk, 0, stream>>>(
        qg, kvb, pos_enc, scale_p, power_p, qp, qn, kp, kn, vh);

    // 4. kmean
    kmean_kernel<<<B_ * H_, 128, 0, stream>>>(kp, kn, kmean);

    // 5. kvm outer products
    kvm_kernel<<<B_ * H_, blk, 0, stream>>>(kp, kn, vh, kvm);

    // 6. x_sim / x_opp with z normalization
    xapply_kernel<<<B_ * H_ * (N_ / 64), blk, 0, stream>>>(qp, qn, kvm, kmean, xo);

    // 7. depthwise conv
    dwconv_kernel<<<B_ * H_ * 64, blk, 0, stream>>>(vh, dwc_w, dwc_b, vc);

    // 8. P = (xo + vc) * g  -> bf16
    combine_pack_kernel<<<(B_ * N_ * C_) / 256, blk, 0, stream>>>(xo, vc, qg, Pbf);

    // 9. out = P @ Wproj + bproj
    gemm_wmma_bf16_async<<<dim3(C_ / 128, M / 128), blk, 0, stream>>>(
        Pbf, WprojT, bproj, out, M, C_, C_);
}
